// AttnNeck_86311662780673
// MI455X (gfx1250) — compile-verified, matrix operand order checked by reading, not run on hardware
//
#include <hip/hip_runtime.h>

typedef __attribute__((ext_vector_type(16))) __bf16 bf16x16;
typedef __attribute__((ext_vector_type(8)))  float  f32x8;

#define BQ    2
#define CC    256
#define HWSZ  4096
#define KCONV 2304

__device__ __forceinline__ unsigned short f2bf(float f) {
  union { float f; unsigned u; } v; v.f = f;
  unsigned r = v.u + 0x7FFFu + ((v.u >> 16) & 1u);   // round-to-nearest-even
  return (unsigned short)(r >> 16);
}

union FragU { uint4 q[2]; bf16x16 v; };

// LDS tile is [128 rows][32 K] bf16, row-major. Per the CDNA5 16-bit A/B layout:
// lane<16 : row = base+lane,    elems 0..7 = K+0..7,  elems 8..15 = K+16..23
// lane>=16: row = base+lane-16, elems 0..7 = K+8..15, elems 8..15 = K+24..31
__device__ __forceinline__ bf16x16 load_frag(const unsigned short* tile, int rowbase, int lane) {
  int r  = rowbase + (lane & 15);
  int kh = (lane >> 4) << 3;
  FragU f;
  f.q[0] = *(const uint4*)(tile + r * 32 + kh);
  f.q[1] = *(const uint4*)(tile + r * 32 + 16 + kh);
  return f.v;
}

// [B][C][HW] f32 -> [B][HW][C] bf16
__global__ __launch_bounds__(256) void cvt_act_kernel(const float* __restrict__ in,
                                                      unsigned short* __restrict__ out) {
  long long o = (long long)blockIdx.x * 256 + threadIdx.x;  // B*HW*C = 2^21
  int c  = (int)(o & 255);
  int hw = (int)((o >> 8) & 4095);
  int b  = (int)(o >> 20);
  out[o] = f2bf(in[(((long long)b * CC + c) << 12) + hw]);
}

// [O][I][3][3] f32 -> [O][tap*256+I] bf16
__global__ __launch_bounds__(256) void cvt_w_kernel(const float* __restrict__ w,
                                                    unsigned short* __restrict__ wp) {
  int idx = blockIdx.x * 256 + threadIdx.x;  // 256*2304
  int co  = idx / KCONV;
  int r   = idx - co * KCONV;
  int tap = r >> 8;
  int ci  = r & 255;
  wp[idx] = f2bf(w[(co * 256 + ci) * 9 + tap]);
}

// Implicit-GEMM conv: M=HW (A = shifted input), N=C_out (Bt = packed weights), K=2304
__global__ __launch_bounds__(256) void conv_gemm_kernel(
    const unsigned short* __restrict__ inp,  // [B][HW][C] bf16
    const unsigned short* __restrict__ wp,   // [256][2304] bf16
    const float* __restrict__ bias,
    unsigned short* __restrict__ outp, int transposeStore) {
  __shared__ __align__(16) unsigned short As[2][128 * 32];
  __shared__ __align__(16) unsigned short Bs[2][128 * 32];
  const int b     = blockIdx.z;
  const int Mbase = blockIdx.y * 128;  // spatial tile
  const int Nbase = blockIdx.x * 128;  // c_out tile
  const int t = threadIdx.x, lane = t & 31, wave = t >> 5;
  const int srow = t >> 1, half = t & 1;
  const int n_s = Mbase + srow, h_s = n_s >> 6, w_s = n_s & 63;
  const unsigned short* inB  = inp + (size_t)b * HWSZ * CC;
  const unsigned short* wrow = wp + (size_t)(Nbase + srow) * KCONV + half * 16;
  f32x8 acc[8] = {};

  auto stage = [&](int buf, int tap, int cb) {
    const int hh = h_s + tap / 3 - 1, ww = w_s + tap % 3 - 1;
    uint4 a0 = make_uint4(0u, 0u, 0u, 0u), a1 = a0;
    if ((unsigned)hh < 64u && (unsigned)ww < 64u) {
      const unsigned short* src = inB + (size_t)(hh * 64 + ww) * CC + cb + half * 16;
      a0 = *(const uint4*)src;
      a1 = *(const uint4*)(src + 8);
    }
    *(uint4*)&As[buf][srow * 32 + half * 16]     = a0;
    *(uint4*)&As[buf][srow * 32 + half * 16 + 8] = a1;
    const int kb = tap * 256 + cb;
    *(uint4*)&Bs[buf][srow * 32 + half * 16]     = *(const uint4*)(wrow + kb);
    *(uint4*)&Bs[buf][srow * 32 + half * 16 + 8] = *(const uint4*)(wrow + kb + 8);
  };
  auto compute = [&](int buf) {
    bf16x16 af = load_frag(&As[buf][0], wave * 16, lane);
    bf16x16 bf[8];
#pragma unroll
    for (int j = 0; j < 8; ++j) bf[j] = load_frag(&Bs[buf][0], j * 16, lane);
#pragma unroll
    for (int j = 0; j < 8; ++j)
      acc[j] = __builtin_amdgcn_wmma_f32_16x16x32_bf16(false, af, false, bf[j],
                                                       (short)0, acc[j], false, false);
  };

  stage(0, 0, 0);
  __syncthreads();
  int idx = 0;
  for (int tap = 0; tap < 9; ++tap) {
    for (int cb = 0; cb < 256; cb += 32, ++idx) {
      const int buf = idx & 1;
      int ntap = tap, ncb = cb + 32;
      if (ncb == 256) { ntap = tap + 1; ncb = 0; }
      if (ntap < 9) stage(buf ^ 1, ntap, ncb);
      compute(buf);
      __syncthreads();
    }
  }

  const int rbase = Mbase + wave * 16 + ((lane >> 4) << 3);
  const int cl = lane & 15;
#pragma unroll
  for (int j = 0; j < 8; ++j) {
    int co = Nbase + j * 16 + cl;
    float bv = bias[co];
#pragma unroll
    for (int i = 0; i < 8; ++i) {
      int n = rbase + i;
      float v = acc[j][i] + bv;
      v = v > 0.f ? v : 0.f;
      size_t o = transposeStore ? (size_t)(b * CC + co) * HWSZ + n
                                : (size_t)(b * HWSZ + n) * CC + co;
      outp[o] = f2bf(v);
    }
  }
}

// Generic GEMM: A [M][K] row-major, Bt [N][K] row-major, D [M][N] row-major.
// MODE 0: D = A*B (f32 out, corrT). MODE 1: D = (A*B)*gamma + xres (final).
template <int MODE>
__global__ __launch_bounds__(256) void gemm_kernel(
    const unsigned short* __restrict__ A, const unsigned short* __restrict__ Bt, int K,
    float* __restrict__ corrOut, const float* __restrict__ xres,
    const float* __restrict__ gptr, float* __restrict__ dout, int N) {
  __shared__ __align__(16) unsigned short As[2][128 * 32];
  __shared__ __align__(16) unsigned short Bs[2][128 * 32];
  const int Mbase = blockIdx.y * 128;
  const int Nbase = blockIdx.x * 128;
  const int t = threadIdx.x, lane = t & 31, wave = t >> 5;
  const int srow = t >> 1, half = t & 1;
  const unsigned short* Ar = A + (size_t)(Mbase + srow) * K + half * 16;
  const unsigned short* Br = Bt + (size_t)(Nbase + srow) * K + half * 16;
  f32x8 acc[8] = {};

  auto stage = [&](int buf, int kb) {
    *(uint4*)&As[buf][srow * 32 + half * 16]     = *(const uint4*)(Ar + kb);
    *(uint4*)&As[buf][srow * 32 + half * 16 + 8] = *(const uint4*)(Ar + kb + 8);
    *(uint4*)&Bs[buf][srow * 32 + half * 16]     = *(const uint4*)(Br + kb);
    *(uint4*)&Bs[buf][srow * 32 + half * 16 + 8] = *(const uint4*)(Br + kb + 8);
  };

  stage(0, 0);
  __syncthreads();
  for (int kb = 0; kb < K; kb += 32) {
    const int buf = (kb >> 5) & 1;
    if (kb + 32 < K) {
      stage(buf ^ 1, kb + 32);
      if (kb + 64 < K) {
        __builtin_prefetch(Ar + kb + 64, 0, 1);
        __builtin_prefetch(Br + kb + 64, 0, 1);
      }
    }
    bf16x16 af = load_frag(&As[buf][0], wave * 16, lane);
    bf16x16 bf[8];
#pragma unroll
    for (int j = 0; j < 8; ++j) bf[j] = load_frag(&Bs[buf][0], j * 16, lane);
#pragma unroll
    for (int j = 0; j < 8; ++j)
      acc[j] = __builtin_amdgcn_wmma_f32_16x16x32_bf16(false, af, false, bf[j],
                                                       (short)0, acc[j], false, false);
    __syncthreads();
  }

  const int rbase = Mbase + wave * 16 + ((lane >> 4) << 3);
  const int cl = lane & 15;
  float g = (MODE == 1) ? gptr[0] : 0.f;
#pragma unroll
  for (int j = 0; j < 8; ++j) {
    int c = Nbase + j * 16 + cl;
#pragma unroll
    for (int i = 0; i < 8; ++i) {
      size_t o = (size_t)(rbase + i) * N + c;
      float v = acc[j][i];
      if (MODE == 0) corrOut[o] = v;
      else           dout[o] = v * g + xres[o];
    }
  }
}

// Row-wise softmax on corrT [4096][4096] f32 (row m contiguous over n), bf16 out.
__global__ __launch_bounds__(256) void softmax_kernel(const float* __restrict__ corrT,
                                                      unsigned short* __restrict__ PT) {
  const int m = blockIdx.x;
  const float* row = corrT + (size_t)m * HWSZ;
  __shared__ float red[256];
  const int t = threadIdx.x;
  float mx = -3.0e38f;
  for (int i = t; i < HWSZ; i += 256) mx = fmaxf(mx, row[i]);
  red[t] = mx; __syncthreads();
  for (int s = 128; s > 0; s >>= 1) { if (t < s) red[t] = fmaxf(red[t], red[t + s]); __syncthreads(); }
  mx = red[0]; __syncthreads();
  float sum = 0.f;
  for (int i = t; i < HWSZ; i += 256) sum += __expf(row[i] - mx);
  red[t] = sum; __syncthreads();
  for (int s = 128; s > 0; s >>= 1) { if (t < s) red[t] += red[t + s]; __syncthreads(); }
  float inv = 1.0f / red[0];
  unsigned short* prow = PT + (size_t)m * HWSZ;
  for (int i = t; i < HWSZ; i += 256) prow[i] = f2bf(__expf(row[i] - mx) * inv);
}

extern "C" void kernel_launch(void* const* d_in, const int* in_sizes, int n_in,
                              void* d_out, int out_size, void* d_ws, size_t ws_size,
                              hipStream_t stream) {
  const float* x     = (const float*)d_in[0];
  const float* rx    = (const float*)d_in[1];
  const float* w1    = (const float*)d_in[2];
  const float* b1    = (const float*)d_in[3];
  const float* w2    = (const float*)d_in[4];
  const float* b2    = (const float*)d_in[5];
  const float* gamma = (const float*)d_in[6];
  float* out = (float*)d_out;

  char* p = (char*)d_ws;
  auto take = [&](size_t bytes) { char* r = p; p += (bytes + 255) & ~(size_t)255; return r; };
  unsigned short* xb    = (unsigned short*)take((size_t)BQ * HWSZ * CC * 2);
  unsigned short* rb    = (unsigned short*)take((size_t)BQ * HWSZ * CC * 2);
  unsigned short* wp1   = (unsigned short*)take((size_t)CC * KCONV * 2);
  unsigned short* wp2   = (unsigned short*)take((size_t)CC * KCONV * 2);
  unsigned short* xfT   = (unsigned short*)take((size_t)BQ * HWSZ * CC * 2);  // [B][HW][C]
  unsigned short* rf    = (unsigned short*)take((size_t)BQ * HWSZ * CC * 2);  // [B][HW][C]
  unsigned short* c2    = (unsigned short*)take((size_t)BQ * CC * HWSZ * 2);  // [B][C][HW]
  float*          corrT = (float*)take((size_t)HWSZ * HWSZ * 4);              // per batch
  unsigned short* PT    = (unsigned short*)take((size_t)HWSZ * HWSZ * 2);     // per batch

  cvt_act_kernel<<<8192, 256, 0, stream>>>(x, xb);
  cvt_act_kernel<<<8192, 256, 0, stream>>>(rx, rb);
  cvt_w_kernel<<<2304, 256, 0, stream>>>(w1, wp1);
  cvt_w_kernel<<<2304, 256, 0, stream>>>(w2, wp2);

  dim3 cg(2, 32, 2);  // N=C_out/128, M=HW/128, B
  conv_gemm_kernel<<<cg, 256, 0, stream>>>(xb, wp1, b1, xfT, 0);
  conv_gemm_kernel<<<cg, 256, 0, stream>>>(rb, wp1, b1, rf, 0);
  conv_gemm_kernel<<<cg, 256, 0, stream>>>(xb, wp2, b2, c2, 1);

  for (int b = 0; b < BQ; ++b) {
    // corrT[m][n] = sum_c xfT[m][c] * rf[n][c]
    gemm_kernel<0><<<dim3(32, 32, 1), 256, 0, stream>>>(
        xfT + (size_t)b * HWSZ * CC, rf + (size_t)b * HWSZ * CC, 256,
        corrT, nullptr, nullptr, nullptr, HWSZ);
    softmax_kernel<<<HWSZ, 256, 0, stream>>>(corrT, PT);
    // out[c][m] = gamma * sum_n c2[c][n] * PT[m][n] + x[c][m]
    gemm_kernel<1><<<dim3(32, 2, 1), 256, 0, stream>>>(
        c2 + (size_t)b * CC * HWSZ, PT, HWSZ,
        nullptr, x + (size_t)b * CC * HWSZ, gamma, out + (size_t)b * CC * HWSZ, HWSZ);
  }
}
